// Transformer_Light_27350351741676
// MI455X (gfx1250) — matrix operand order repaired and checked by reading, not compiled, not run on hardware
//
#include <hip/hip_runtime.h>
#include <hip/hip_bf16.h>

// ---------------------------------------------------------------------------
// Problem constants (from reference)
// ---------------------------------------------------------------------------
#define B_   8
#define P_   64
#define T_   63
#define K_   2
#define D_   512
#define DFF_ 2048
#define L_   2
#define V_   30522
#define NC_  18
#define DPT_ 1536
#define DWD_ 768
#define S_   127          // P + T

typedef __bf16 bf16_t;
typedef __attribute__((ext_vector_type(16))) bf16_t v16bf;
typedef __attribute__((ext_vector_type(8)))  bf16_t v8bf;
typedef __attribute__((ext_vector_type(4)))  bf16_t v4bf;
typedef __attribute__((ext_vector_type(2)))  bf16_t v2bf;
typedef __attribute__((ext_vector_type(8)))  float  v8f;
typedef __attribute__((ext_vector_type(4)))  float  v4f;

__device__ __forceinline__ bf16_t f2bf(float f) { return (bf16_t)f; }

union FragU { v16bf v; v8bf h[2]; };

// ---------------------------------------------------------------------------
// Generic batched GEMM:  C[b] = epilogue( A[b](MxK) * B[b](KxN) )
//   fp32 in global, bf16 staged in LDS, v_wmma_f32_16x16x32_bf16 accumulate.
//   Block tile 64x64, K-step 64 (2 WMMA k-substeps), 128 threads = 4 wave32
//   waves, each wave owns a 32x32 quadrant (2x2 accumulators, 8 WMMA/iter).
//   B is stored TRANSPOSED in LDS (Bs[n][k]) so A and B fragments are
//   contiguous 16B chunks -> ds_load_b128 pairs.
//   Interior tiles: branchless float4 staging (global_load_b128 + packed cvt
//   + ds_store_b64). Edge tiles: clamp+select, no exec branching on loads.
// ---------------------------------------------------------------------------
#define BM 64
#define BN 64
#define BK 64

template<int TRANSB, int HASBIAS, int HASRES, int RELU>
__global__ __launch_bounds__(128) void gemm_wmma_kernel(
    const float* __restrict__ A, long aBatch, int lda,
    const float* __restrict__ Bm, long bBatch, int ldb,
    const float* __restrict__ bias,
    const float* __restrict__ resid, long rBatch, int ldr,
    float* __restrict__ C, long cBatch, int ldc,
    int M, int N, int Kd)
{
    __shared__ bf16_t As[BM][BK + 8];
    __shared__ bf16_t Bs[BN][BK + 8];   // transposed: Bs[n][k]

    const int batch = blockIdx.z;
    const float* Ab = A  + (long)batch * aBatch;
    const float* Bb = Bm + (long)batch * bBatch;
    float*       Cb = C  + (long)batch * cBatch;

    const int m0 = blockIdx.y * BM;
    const int n0 = blockIdx.x * BN;
    const int tid  = threadIdx.x;         // 0..127
    const int wave = tid >> 5;            // 0..3 (wave32)
    const int lane = tid & 31;
    const int kh   = lane >> 4;           // K-half selector per ISA layout
    const int rl   = lane & 15;

    const int wr = (wave >> 1) * 32;      // wave row offset in block tile
    const int wc = (wave & 1) * 32;       // wave col offset in block tile

    const bool mFull = (m0 + BM <= M);
    const bool nFull = (n0 + BN <= N);

    v8f zero = {0.f,0.f,0.f,0.f,0.f,0.f,0.f,0.f};
    v8f acc[2][2];
    acc[0][0] = zero; acc[0][1] = zero; acc[1][0] = zero; acc[1][1] = zero;

    for (int k0 = 0; k0 < Kd; k0 += BK) {
        const bool kFull = (k0 + BK <= Kd);

        // ---- stage A tile (BM x BK) ----
        if (mFull && kFull) {
            // 4096 elems / 128 thr = 8 float4 quads each
            #pragma unroll
            for (int i = 0; i < 8; ++i) {
                int idx = tid + i * 128;              // 0..1023 quads
                int r = idx >> 4, c = (idx & 15) << 2;
                const v4f q = *(const v4f*)(Ab + (long)(m0 + r) * lda + (k0 + c));
                *(v4bf*)&As[r][c] = (v4bf){ f2bf(q.x), f2bf(q.y), f2bf(q.z), f2bf(q.w) };
            }
        } else {
            // 2048 pairs, clamp + select
            #pragma unroll
            for (int i = 0; i < 16; ++i) {
                int idx = tid + i * 128;
                int r = idx >> 5, c = (idx & 31) << 1;
                int gr = m0 + r, gc = k0 + c;
                int grc = gr < M ? gr : (M - 1);
                int g0 = gc < Kd ? gc : (Kd - 1);
                int g1 = (gc + 1) < Kd ? (gc + 1) : (Kd - 1);
                const float* ap = Ab + (long)grc * lda;
                float v0 = ap[g0], v1 = ap[g1];
                v0 = (gr < M && gc     < Kd) ? v0 : 0.f;
                v1 = (gr < M && gc + 1 < Kd) ? v1 : 0.f;
                *(v2bf*)&As[r][c] = (v2bf){ f2bf(v0), f2bf(v1) };
            }
        }

        // ---- stage B tile into transposed LDS (Bs[n][k]) ----
        if (TRANSB) {
            // global B is (N x K): quads along k are contiguous
            if (nFull && kFull) {
                #pragma unroll
                for (int i = 0; i < 8; ++i) {
                    int idx = tid + i * 128;
                    int n = idx >> 4, c = (idx & 15) << 2;
                    const v4f q = *(const v4f*)(Bb + (long)(n0 + n) * ldb + (k0 + c));
                    *(v4bf*)&Bs[n][c] = (v4bf){ f2bf(q.x), f2bf(q.y), f2bf(q.z), f2bf(q.w) };
                }
            } else {
                #pragma unroll
                for (int i = 0; i < 16; ++i) {
                    int idx = tid + i * 128;
                    int n = idx >> 5, c = (idx & 31) << 1;
                    int gn = n0 + n, gc = k0 + c;
                    int gnc = gn < N ? gn : (N - 1);
                    int g0 = gc < Kd ? gc : (Kd - 1);
                    int g1 = (gc + 1) < Kd ? (gc + 1) : (Kd - 1);
                    const float* bp = Bb + (long)gnc * ldb;
                    float v0 = bp[g0], v1 = bp[g1];
                    v0 = (gn < N && gc     < Kd) ? v0 : 0.f;
                    v1 = (gn < N && gc + 1 < Kd) ? v1 : 0.f;
                    *(v2bf*)&Bs[n][c] = (v2bf){ f2bf(v0), f2bf(v1) };
                }
            }
        } else {
            // global B is (K x N): coalesce across n, pair along k
            if (nFull && kFull) {
                #pragma unroll
                for (int i = 0; i < 16; ++i) {
                    int idx = tid + i * 128;          // 0..2047 pairs
                    int n = idx & 63, c = (idx >> 6) << 1;
                    const float* bp = Bb + (long)(k0 + c) * ldb + (n0 + n);
                    *(v2bf*)&Bs[n][c] = (v2bf){ f2bf(bp[0]), f2bf(bp[ldb]) };
                }
            } else {
                #pragma unroll
                for (int i = 0; i < 16; ++i) {
                    int idx = tid + i * 128;
                    int n = idx & 63, c = (idx >> 6) << 1;
                    int gn = n0 + n, gc = k0 + c;
                    int gnc = gn < N ? gn : (N - 1);
                    int g0 = gc < Kd ? gc : (Kd - 1);
                    int g1 = (gc + 1) < Kd ? (gc + 1) : (Kd - 1);
                    float v0 = Bb[(long)g0 * ldb + gnc];
                    float v1 = Bb[(long)g1 * ldb + gnc];
                    v0 = (gn < N && gc     < Kd) ? v0 : 0.f;
                    v1 = (gn < N && gc + 1 < Kd) ? v1 : 0.f;
                    *(v2bf*)&Bs[n][c] = (v2bf){ f2bf(v0), f2bf(v1) };
                }
            }
        }

        // prefetch next K tile (speculative; global_prefetch_b8)
        if (k0 + BK < Kd) {
            int pr = m0 + (tid >> 2); if (pr >= M) pr = M - 1;
            __builtin_prefetch(&Ab[(long)pr * lda + (k0 + BK)], 0, 1);
        }
        __syncthreads();

        // ---- 2 K-substeps x (2x2) WMMAs; fragments are 16B LDS chunks ----
        #pragma unroll
        for (int ks = 0; ks < 2; ++ks) {
            const int kb = ks * 32;
            FragU afrag[2], bfrag[2];
            #pragma unroll
            for (int mt = 0; mt < 2; ++mt) {
                int row = wr + mt * 16 + rl;
                afrag[mt].h[0] = *(const v8bf*)&As[row][kb + kh * 8];
                afrag[mt].h[1] = *(const v8bf*)&As[row][kb + 16 + kh * 8];
            }
            #pragma unroll
            for (int nt = 0; nt < 2; ++nt) {
                int col = wc + nt * 16 + rl;
                bfrag[nt].h[0] = *(const v8bf*)&Bs[col][kb + kh * 8];
                bfrag[nt].h[1] = *(const v8bf*)&Bs[col][kb + 16 + kh * 8];
            }
            #pragma unroll
            for (int mt = 0; mt < 2; ++mt)
                #pragma unroll
                for (int nt = 0; nt < 2; ++nt)
                    acc[mt][nt] = __builtin_amdgcn_wmma_f32_16x16x32_bf16(
                        false, afrag[mt].v, false, bfrag[nt].v,
                        (short)0, acc[mt][nt], false, false);
        }
        __syncthreads();
    }

    // ---- epilogue ----
    const float* Rb = HASRES ? (resid + (long)batch * rBatch) : nullptr;
    if (mFull && nFull) {
        #pragma unroll
        for (int mt = 0; mt < 2; ++mt)
            #pragma unroll
            for (int nt = 0; nt < 2; ++nt) {
                int col = n0 + wc + nt * 16 + rl;
                #pragma unroll
                for (int r = 0; r < 8; ++r) {
                    int row = m0 + wr + mt * 16 + r + kh * 8;
                    float v = acc[mt][nt][r];
                    if (HASBIAS) v += bias[col];
                    if (HASRES)  v += Rb[(long)row * ldr + col];
                    if (RELU)    v = fmaxf(v, 0.f);
                    Cb[(long)row * ldc + col] = v;
                }
            }
    } else {
        #pragma unroll
        for (int mt = 0; mt < 2; ++mt)
            #pragma unroll
            for (int nt = 0; nt < 2; ++nt) {
                int col = n0 + wc + nt * 16 + rl;
                #pragma unroll
                for (int r = 0; r < 8; ++r) {
                    int row = m0 + wr + mt * 16 + r + kh * 8;
                    if (row < M && col < N) {
                        float v = acc[mt][nt][r];
                        if (HASBIAS) v += bias[col];
                        if (HASRES)  v += Rb[(long)row * ldr + col];
                        if (RELU)    v = fmaxf(v, 0.f);
                        Cb[(long)row * ldc + col] = v;
                    }
                }
            }
    }
}

// ---------------------------------------------------------------------------
// Attention softmax over one row of scores (cols = 127), optional seq2seq
// mask: allowed = (j<P) | (i>=P & j<=i).  grid: (S_, nseq), block: 128
// ---------------------------------------------------------------------------
__global__ void attn_softmax_kernel(float* __restrict__ S, int ld, int cols,
                                    int masked, float scale)
{
    int i   = blockIdx.x;
    int seq = blockIdx.y;
    float* row = S + ((long)seq * S_ + i) * ld;
    int t = threadIdx.x;
    __shared__ float red[128];

    float v = -1e30f;
    if (t < cols) {
        float mb = 0.f;
        if (masked) {
            bool ok = (t < P_) || (i >= P_ && t <= i);
            if (!ok) mb = -1e9f;
        }
        v = row[t] * scale + mb;
    }
    red[t] = v; __syncthreads();
    for (int off = 64; off > 0; off >>= 1) {
        if (t < off) red[t] = fmaxf(red[t], red[t + off]);
        __syncthreads();
    }
    float mx = red[0]; __syncthreads();
    float e = (t < cols) ? expf(v - mx) : 0.f;
    red[t] = e; __syncthreads();
    for (int off = 64; off > 0; off >>= 1) {
        if (t < off) red[t] += red[t + off];
        __syncthreads();
    }
    float inv = 1.f / red[0];
    if (t < cols) row[t] = e * inv;
}

// ---------------------------------------------------------------------------
// Post-norm LayerNorm over dim=512; grid = ntok, block = 256
// ---------------------------------------------------------------------------
__global__ void layernorm_kernel(const float* __restrict__ in, float* __restrict__ out,
                                 const float* __restrict__ g, const float* __restrict__ b,
                                 int dim)
{
    long row = blockIdx.x;
    const float* x = in + row * dim;
    float* o = out + row * dim;
    __shared__ float rs[256], rq[256];
    int t = threadIdx.x;
    float s = 0.f, q = 0.f;
    for (int d = t; d < dim; d += 256) { float v = x[d]; s += v; q += v * v; }
    rs[t] = s; rq[t] = q; __syncthreads();
    for (int off = 128; off > 0; off >>= 1) {
        if (t < off) { rs[t] += rs[t + off]; rq[t] += rq[t + off]; }
        __syncthreads();
    }
    float mean = rs[0] / dim;
    float var  = rq[0] / dim - mean * mean;
    float inv  = rsqrtf(var + 1e-5f);
    for (int d = t; d < dim; d += 256)
        o[d] = (x[d] - mean) * inv * g[d] + b[d];
}

// ---------------------------------------------------------------------------
// mean over proposals; grid B_, block 256
// ---------------------------------------------------------------------------
__global__ void mean_box_kernel(const float* __restrict__ box, float* __restrict__ mean)
{
    int b = blockIdx.x;
    for (int d = threadIdx.x; d < D_; d += 256) {
        float s = 0.f;
        for (int p = 0; p < P_; ++p) s += box[((long)(b * P_ + p)) * D_ + d];
        mean[b * D_ + d] = s * (1.f / P_);
    }
}

// VG sequence assembly: grid (S_, B_), block 256
__global__ void build_vg_kernel(const float* __restrict__ box, const float* __restrict__ txt,
                                const float* __restrict__ mean, float* __restrict__ x)
{
    int tok = blockIdx.x, b = blockIdx.y;
    float* dst = x + ((long)(b * S_ + tok)) * D_;
    for (int d = threadIdx.x; d < D_; d += 256) {
        float v;
        if (tok < P_) v = box[((long)(b * P_ + tok)) * D_ + d];
        else          v = txt[((long)(b * T_ + tok - P_)) * D_ + d] + mean[b * D_ + d];
        dst[d] = v;
    }
}

// DC sequence assembly: grid (S_, B_*K_), block 256
__global__ void build_dc_kernel(const float* __restrict__ box, const float* __restrict__ txt,
                                const int* __restrict__ tp, float* __restrict__ x)
{
    int tok = blockIdx.x, s = blockIdx.y;
    int b = s / K_, k = s % K_;
    float* dst = x + ((long)(s * S_ + tok)) * D_;
    int prop = tp[b * K_ + k];
    for (int d = threadIdx.x; d < D_; d += 256) {
        float v;
        if (tok < P_) v = box[((long)(b * P_ + tok)) * D_ + d];
        else          v = txt[((long)(b * T_ + tok - P_)) * D_ + d]
                        + box[((long)(b * P_ + prop)) * D_ + d];
        dst[d] = v;
    }
}

// gather box tokens + CLS token from encoded VG: grid (P_+1, B_), block 256
__global__ void gather_vg_kernel(const float* __restrict__ xvg,
                                 float* __restrict__ boxtok, float* __restrict__ clstok)
{
    int tok = blockIdx.x, b = blockIdx.y;
    const float* src = xvg + ((long)(b * S_ + tok)) * D_;
    float* dst = (tok < P_) ? (boxtok + ((long)(b * P_ + tok)) * D_)
                            : (clstok + (long)b * D_);
    for (int d = threadIdx.x; d < D_; d += 256) dst[d] = src[d];
}

// gather text tokens from encoded DC: grid (T_, B_*K_), block 256
__global__ void gather_txt_kernel(const float* __restrict__ xdc, float* __restrict__ capin)
{
    int t = blockIdx.x, s = blockIdx.y;
    const float* src = xdc + ((long)(s * S_ + P_ + t)) * D_;
    float* dst = capin + ((long)(s * T_ + t)) * D_;
    for (int d = threadIdx.x; d < D_; d += 256) dst[d] = src[d];
}

// ---------------------------------------------------------------------------
// Losses
// ---------------------------------------------------------------------------
__global__ void match_loss_kernel(const float* __restrict__ scores,
                                  const int* __restrict__ tp, float* __restrict__ out)
{
    __shared__ float lb[B_];
    int b = threadIdx.x;
    if (b < B_) {
        const float* s = scores + b * P_;
        float mx = -1e30f;
        for (int p = 0; p < P_; ++p) mx = fmaxf(mx, s[p]);
        float sum = 0.f;
        for (int p = 0; p < P_; ++p) sum += expf(s[p] - mx);
        float lse = mx + logf(sum);
        int t0 = tp[b * K_], t1 = tp[b * K_ + 1];
        float loss = 0.f;
        for (int p = 0; p < P_; ++p)
            if (p == t0 || p == t1) loss += (1.f / K_) * (lse - s[p]);
        lb[b] = loss;
    }
    __syncthreads();
    if (threadIdx.x == 0) {
        float m = 0.f;
        for (int b2 = 0; b2 < B_; ++b2) m += lb[b2];
        out[0] = m / B_;
    }
}

__global__ void cls_loss_kernel(const float* __restrict__ scores,
                                const int* __restrict__ tc, float* __restrict__ out)
{
    __shared__ float lb[B_];
    int b = threadIdx.x;
    if (b < B_) {
        const float* s = scores + b * NC_;
        float mx = -1e30f;
        for (int c = 0; c < NC_; ++c) mx = fmaxf(mx, s[c]);
        float sum = 0.f;
        for (int c = 0; c < NC_; ++c) sum += expf(s[c] - mx);
        float lse = mx + logf(sum);
        lb[b] = lse - s[tc[b]];
    }
    __syncthreads();
    if (threadIdx.x == 0) {
        float m = 0.f;
        for (int b2 = 0; b2 < B_; ++b2) m += lb[b2];
        out[0] = m / B_;
    }
}

// per-row -log_softmax[wid] over V_; grid B_*K_*T_, block 256
__global__ void dc_rowloss_kernel(const float* __restrict__ logits,
                                  const int* __restrict__ wids,
                                  float* __restrict__ rowloss)
{
    long row = blockIdx.x;
    const float* s = logits + row * (long)V_;
    __shared__ float red[256];
    int t = threadIdx.x;
    float mx = -1e30f;
    for (int j = t; j < V_; j += 256) mx = fmaxf(mx, s[j]);
    red[t] = mx; __syncthreads();
    for (int off = 128; off > 0; off >>= 1) {
        if (t < off) red[t] = fmaxf(red[t], red[t + off]);
        __syncthreads();
    }
    mx = red[0]; __syncthreads();
    float sm = 0.f;
    for (int j = t; j < V_; j += 256) sm += expf(s[j] - mx);
    red[t] = sm; __syncthreads();
    for (int off = 128; off > 0; off >>= 1) {
        if (t < off) red[t] += red[t + off];
        __syncthreads();
    }
    if (t == 0) {
        float lse = mx + logf(red[0]);
        int b  = (int)(row / (K_ * T_));
        int tt = (int)(row % T_);
        int w = wids[b * (T_ + 1) + 1 + tt];
        rowloss[row] = lse - s[w];
    }
}

__global__ void dc_loss_kernel(const float* __restrict__ rowloss, float* __restrict__ out)
{
    __shared__ float red[256];
    int t = threadIdx.x;
    float s = 0.f;
    for (int i = t; i < B_ * K_ * T_; i += 256) s += rowloss[i];
    red[t] = s; __syncthreads();
    for (int off = 128; off > 0; off >>= 1) {
        if (t < off) red[t] += red[t + off];
        __syncthreads();
    }
    if (t == 0) out[0] = red[0] / (float)(T_ * B_);   // sum over K, mean over T,B
}

// ---------------------------------------------------------------------------
// Host-side helpers
// ---------------------------------------------------------------------------
template<int TB, int HB, int HR, int RL>
static inline void gemm_t(hipStream_t st,
                          const float* A, long aB, int lda,
                          const float* Bm, long bB, int ldb,
                          const float* bias, const float* resid, long rB, int ldr,
                          float* C, long cB, int ldc,
                          int M, int N, int K, int batch)
{
    dim3 g((N + BN - 1) / BN, (M + BM - 1) / BM, batch), blk(128);
    gemm_wmma_kernel<TB, HB, HR, RL><<<g, blk, 0, st>>>(
        A, aB, lda, Bm, bB, ldb, bias, resid, rB, ldr, C, cB, ldc, M, N, K);
}

static void gemm(hipStream_t st,
                 const float* A, long aB, int lda,
                 const float* Bm, long bB, int ldb, int transB,
                 const float* bias, const float* resid, long rB, int ldr,
                 float* C, long cB, int ldc,
                 int M, int N, int K, int relu, int batch)
{
    if (transB)
        gemm_t<1,0,0,0>(st, A,aB,lda, Bm,bB,ldb, nullptr, nullptr,0,0, C,cB,ldc, M,N,K, batch);
    else if (resid)
        gemm_t<0,1,1,0>(st, A,aB,lda, Bm,bB,ldb, bias, resid,rB,ldr, C,cB,ldc, M,N,K, batch);
    else if (relu)
        gemm_t<0,1,0,1>(st, A,aB,lda, Bm,bB,ldb, bias, nullptr,0,0, C,cB,ldc, M,N,K, batch);
    else if (bias)
        gemm_t<0,1,0,0>(st, A,aB,lda, Bm,bB,ldb, bias, nullptr,0,0, C,cB,ldc, M,N,K, batch);
    else
        gemm_t<0,0,0,0>(st, A,aB,lda, Bm,bB,ldb, nullptr, nullptr,0,0, C,cB,ldc, M,N,K, batch);
}

static void encode_branch(float* x, int nseq, int masked,
                          const float* qkv_w, const float* qkv_b,
                          const float* out_w, const float* out_b,
                          const float* ln1_g, const float* ln1_b,
                          const float* ff1_w, const float* ff1_b,
                          const float* ff2_w, const float* ff2_b,
                          const float* ln2_g, const float* ln2_b,
                          float* qkv, float* scores, float* attn,
                          float* pre, float* ffh, hipStream_t st)
{
    const int Ntok = nseq * S_;
    const float scale = 0.044194173824159216f;   // 1/sqrt(512)
    for (int l = 0; l < L_; ++l) {
        // QKV projection over all tokens
        gemm(st, x, 0, D_, qkv_w + (long)l * D_ * 3 * D_, 0, 3 * D_, 0,
             qkv_b + (long)l * 3 * D_, nullptr, 0, 0,
             qkv, 0, 3 * D_, Ntok, 3 * D_, D_, 0, 1);
        // scores = q @ k^T  (batched, B transposed)
        gemm(st, qkv, (long)S_ * 3 * D_, 3 * D_,
             qkv + D_, (long)S_ * 3 * D_, 3 * D_, 1,
             nullptr, nullptr, 0, 0,
             scores, (long)S_ * 128, 128, S_, S_, D_, 0, nseq);
        attn_softmax_kernel<<<dim3(S_, nseq), 128, 0, st>>>(scores, 128, S_, masked, scale);
        // attn = softmax @ v
        gemm(st, scores, (long)S_ * 128, 128,
             qkv + 2 * D_, (long)S_ * 3 * D_, 3 * D_, 0,
             nullptr, nullptr, 0, 0,
             attn, (long)S_ * D_, D_, S_, D_, S_, 0, nseq);
        // out projection + residual, then LN1
        gemm(st, attn, 0, D_, out_w + (long)l * D_ * D_, 0, D_, 0,
             out_b + (long)l * D_, x, 0, D_,
             pre, 0, D_, Ntok, D_, D_, 0, 1);
        layernorm_kernel<<<Ntok, 256, 0, st>>>(pre, x, ln1_g + l * D_, ln1_b + l * D_, D_);
        // FFN: relu(x@W1+b1)@W2+b2 + residual, then LN2
        gemm(st, x, 0, D_, ff1_w + (long)l * D_ * DFF_, 0, DFF_, 0,
             ff1_b + (long)l * DFF_, nullptr, 0, 0,
             ffh, 0, DFF_, Ntok, DFF_, D_, 1, 1);
        gemm(st, ffh, 0, DFF_, ff2_w + (long)l * DFF_ * D_, 0, D_, 0,
             ff2_b + (long)l * D_, x, 0, D_,
             pre, 0, D_, Ntok, D_, DFF_, 0, 1);
        layernorm_kernel<<<Ntok, 256, 0, st>>>(pre, x, ln2_g + l * D_, ln2_b + l * D_, D_);
    }
}

// ---------------------------------------------------------------------------
// Entry point
// ---------------------------------------------------------------------------
extern "C" void kernel_launch(void* const* d_in, const int* in_sizes, int n_in,
                              void* d_out, int out_size, void* d_ws, size_t ws_size,
                              hipStream_t stream)
{
    (void)in_sizes; (void)n_in; (void)out_size; (void)ws_size;

    const float* instances = (const float*)d_in[0];
    const float* text      = (const float*)d_in[1];
    const int*   tp        = (const int*)d_in[2];
    const int*   wid       = (const int*)d_in[3];
    const int*   tc        = (const int*)d_in[4];
    const float* Wi1 = (const float*)d_in[5];  const float* bi1 = (const float*)d_in[6];
    const float* Wi2 = (const float*)d_in[7];  const float* bi2 = (const float*)d_in[8];
    const float* Ww  = (const float*)d_in[9];  const float* bw  = (const float*)d_in[10];
    const float* qkv_w = (const float*)d_in[11]; const float* qkv_b = (const float*)d_in[12];
    const float* out_w = (const float*)d_in[13]; const float* out_b = (const float*)d_in[14];
    const float* ln1_g = (const float*)d_in[15]; const float* ln1_b = (const float*)d_in[16];
    const float* ff1_w = (const float*)d_in[17]; const float* ff1_b = (const float*)d_in[18];
    const float* ff2_w = (const float*)d_in[19]; const float* ff2_b = (const float*)d_in[20];
    const float* ln2_g = (const float*)d_in[21]; const float* ln2_b = (const float*)d_in[22];
    const float* grd_w1 = (const float*)d_in[23]; const float* grd_b1 = (const float*)d_in[24];
    const float* grd_w2 = (const float*)d_in[25]; const float* grd_b2 = (const float*)d_in[26];
    const float* cls_w1 = (const float*)d_in[27]; const float* cls_b1 = (const float*)d_in[28];
    const float* cls_w2 = (const float*)d_in[29]; const float* cls_b2 = (const float*)d_in[30];
    const float* cap_w1 = (const float*)d_in[31]; const float* cap_b1 = (const float*)d_in[32];
    const float* cap_w2 = (const float*)d_in[33]; const float* cap_b2 = (const float*)d_in[34];

    // ---- workspace carve-up (fp32) ----
    float* p = (float*)d_ws;
    auto alloc = [&](long n) { float* r = p; p += n; return r; };
    float* hidden  = alloc((long)B_ * P_ * 1024);        // 512x1024
    float* box     = alloc((long)B_ * P_ * D_);          // 512x512
    float* txtb    = alloc((long)B_ * T_ * D_);          // 504x512
    float* meanb   = alloc((long)B_ * D_);
    float* xvg     = alloc((long)B_ * S_ * D_);          // 1016x512
    float* xdc     = alloc((long)B_ * K_ * S_ * D_);     // 2032x512
    float* qkvbuf  = alloc((long)B_ * K_ * S_ * 3 * D_); // 2032x1536
    float* scores  = alloc((long)B_ * K_ * S_ * 128);    // 16 x 127 x 128
    float* attn    = alloc((long)B_ * K_ * S_ * D_);
    float* pre     = alloc((long)B_ * K_ * S_ * D_);
    float* ffh     = alloc((long)B_ * K_ * S_ * DFF_);   // 2032x2048
    float* boxtok  = alloc((long)B_ * P_ * D_);
    float* clstok  = alloc((long)B_ * D_);
    float* grdh    = alloc((long)B_ * P_ * 256);
    float* clsh    = alloc((long)B_ * 256);
    float* capin   = alloc((long)B_ * K_ * T_ * D_);     // 1008x512
    float* caph    = alloc((long)B_ * K_ * T_ * DWD_);   // 1008x768
    float* rowloss = alloc((long)B_ * K_ * T_);

    // ---- output carve-up ----
    float* out      = (float*)d_out;
    float* o_match  = out;                                   // [B,P] = 512
    float* o_mloss  = out + 512;                             // 1
    float* o_cls    = out + 513;                             // [B,NC] = 144
    float* o_closs  = out + 657;                             // 1
    float* o_dc     = out + 658;                             // [B,K,T,V]
    float* o_dloss  = out + 658 + (long)B_ * K_ * T_ * V_;   // 1

    // ---- embeddings ----
    gemm(stream, instances, 0, DPT_, Wi1, 0, 1024, 0, bi1, nullptr, 0, 0,
         hidden, 0, 1024, B_ * P_, 1024, DPT_, 0, 1);
    gemm(stream, hidden, 0, 1024, Wi2, 0, D_, 0, bi2, nullptr, 0, 0,
         box, 0, D_, B_ * P_, D_, 1024, 0, 1);
    gemm(stream, text, 0, DWD_, Ww, 0, D_, 0, bw, nullptr, 0, 0,
         txtb, 0, D_, B_ * T_, D_, DWD_, 0, 1);
    mean_box_kernel<<<B_, 256, 0, stream>>>(box, meanb);

    // ---- VG branch (unmasked encoder) ----
    build_vg_kernel<<<dim3(S_, B_), 256, 0, stream>>>(box, txtb, meanb, xvg);
    encode_branch(xvg, B_, 0, qkv_w, qkv_b, out_w, out_b, ln1_g, ln1_b,
                  ff1_w, ff1_b, ff2_w, ff2_b, ln2_g, ln2_b,
                  qkvbuf, scores, attn, pre, ffh, stream);
    gather_vg_kernel<<<dim3(P_ + 1, B_), 256, 0, stream>>>(xvg, boxtok, clstok);
    // grounding head -> match_scores directly into d_out (ldc=1)
    gemm(stream, boxtok, 0, D_, grd_w1, 0, 256, 0, grd_b1, nullptr, 0, 0,
         grdh, 0, 256, B_ * P_, 256, D_, 0, 1);
    gemm(stream, grdh, 0, 256, grd_w2, 0, 1, 0, grd_b2, nullptr, 0, 0,
         o_match, 0, 1, B_ * P_, 1, 256, 0, 1);
    // classification head -> cls_scores directly into d_out (ldc=NC)
    gemm(stream, clstok, 0, D_, cls_w1, 0, 256, 0, cls_b1, nullptr, 0, 0,
         clsh, 0, 256, B_, 256, D_, 0, 1);
    gemm(stream, clsh, 0, 256, cls_w2, 0, NC_, 0, cls_b2, nullptr, 0, 0,
         o_cls, 0, NC_, B_, NC_, 256, 0, 1);
    match_loss_kernel<<<1, 32, 0, stream>>>(o_match, tp, o_mloss);
    cls_loss_kernel<<<1, 32, 0, stream>>>(o_cls, tc, o_closs);

    // ---- DC branch (seq2seq mask) ----
    build_dc_kernel<<<dim3(S_, B_ * K_), 256, 0, stream>>>(box, txtb, tp, xdc);
    encode_branch(xdc, B_ * K_, 1, qkv_w, qkv_b, out_w, out_b, ln1_g, ln1_b,
                  ff1_w, ff1_b, ff2_w, ff2_b, ln2_g, ln2_b,
                  qkvbuf, scores, attn, pre, ffh, stream);
    gather_txt_kernel<<<dim3(T_, B_ * K_), 256, 0, stream>>>(xdc, capin);
    // caption head: (1008x512)@(512x768) then (1008x768)@(768x30522) -> d_out
    gemm(stream, capin, 0, D_, cap_w1, 0, DWD_, 0, cap_b1, nullptr, 0, 0,
         caph, 0, DWD_, B_ * K_ * T_, DWD_, D_, 0, 1);
    gemm(stream, caph, 0, DWD_, cap_w2, 0, V_, 0, cap_b2, nullptr, 0, 0,
         o_dc, 0, V_, B_ * K_ * T_, V_, DWD_, 0, 1);
    dc_rowloss_kernel<<<B_ * K_ * T_, 256, 0, stream>>>(o_dc, wid, rowloss);
    dc_loss_kernel<<<1, 256, 0, stream>>>(rowloss, o_dloss);
}
